// HetAttn_30846455120584
// MI455X (gfx1250) — matrix operand order repaired and checked by reading, not compiled, not run on hardware
//
#include <hip/hip_runtime.h>
#include <hip/hip_bf16.h>
#include <math.h>

typedef __attribute__((ext_vector_type(16))) __bf16 v16bf;
typedef __attribute__((ext_vector_type(8)))  float  v8f;

#define NN 50000
#define NR 3
#define NE 800000
#define DH 128

// ---------------------------------------------------------------- utilities
__global__ void zero_f32(float* __restrict__ p, long long n) {
  long long i = (long long)blockIdx.x * blockDim.x + threadIdx.x;
  if (i < n) p[i] = 0.0f;
}

// Convert an f32 weight [K, 128] (or transposed source for concat_w) into the
// per-lane bf16 B-fragment layout for v_wmma_f32_16x16x32_bf16:
// frag[((kt*8+nt)*32+lane)*16 + e] = W[kt*32 + (lane>>4)*8 + (e&7) + (e>>3)*16][nt*16 + (lane&15)]
__global__ void prep_frag(const float* __restrict__ src, __bf16* __restrict__ dst,
                          int K, int trans) {
  int total = (K >> 5) * 8 * 32 * 16;
  int tid = blockIdx.x * blockDim.x + threadIdx.x;
  if (tid >= total) return;
  int e    = tid & 15;
  int t1   = tid >> 4;
  int lane = t1 & 31;
  int t2   = t1 >> 5;
  int nt   = t2 & 7;
  int kt   = t2 >> 3;
  int k = kt * 32 + (lane >> 4) * 8 + (e & 7) + (e >> 3) * 16;
  int n = nt * 16 + (lane & 15);
  float w = trans ? src[(size_t)n * K + k] : src[(size_t)k * DH + n];
  dst[tid] = (__bf16)w;
}

// ---------------------------------------------------------------- degrees
__global__ void count_deg(const int* __restrict__ edges, float* __restrict__ dOut,
                          float* __restrict__ dIn) {
  long long tid = (long long)blockIdx.x * blockDim.x + threadIdx.x;
  if (tid >= (long long)NR * NE) return;
  int r = (int)(tid / NE);
  int e = (int)(tid - (long long)r * NE);
  int s = edges[(size_t)(r * 2 + 0) * NE + e];
  int d = edges[(size_t)(r * 2 + 1) * NE + e];
  atomicAdd(&dOut[(size_t)r * NN + s], 1.0f);
  atomicAdd(&dIn [(size_t)r * NN + d], 1.0f);
}

__global__ void finalize_deg(float* __restrict__ p, long long n) {
  long long i = (long long)blockIdx.x * blockDim.x + threadIdx.x;
  if (i < n) p[i] = rsqrtf(fmaxf(p[i], 1.0f));
}

// ---------------------------------------------------------------- WMMA GEMM
// C[N,128] = (A * rowScale) @ W ; W pre-fragmented bf16. EP: 0 = none, 1 = tanh.
// Block = 256 threads (8 waves); each wave computes 16 rows x 128 cols.
template <int EP>
__global__ __launch_bounds__(256)
void gemm128_wmma(const float* __restrict__ A, const float* __restrict__ rowScale,
                  const __bf16* __restrict__ frag, float* __restrict__ C, int N) {
  const int wave = threadIdx.x >> 5, lane = threadIdx.x & 31;
  const int h = lane >> 4, rl = lane & 15;
  const int rowbase = (blockIdx.x * 8 + wave) * 16;
  int row = rowbase + rl;
  int rowc = row < N ? row : N - 1;
  float s = rowScale ? rowScale[rowc] : 1.0f;
  const float* arow = A + (size_t)rowc * DH;
  const v16bf* fragv = (const v16bf*)frag;

  v8f acc[8] = {};
  #pragma unroll
  for (int kt = 0; kt < 4; ++kt) {
    int kb = kt * 32 + h * 8;          // ISA 16-bit A 16x32 layout
    float4 a0 = *(const float4*)(arow + kb);
    float4 a1 = *(const float4*)(arow + kb + 4);
    float4 a2 = *(const float4*)(arow + kb + 16);
    float4 a3 = *(const float4*)(arow + kb + 20);
    v16bf a;
    a[0]  = (__bf16)(a0.x * s); a[1]  = (__bf16)(a0.y * s);
    a[2]  = (__bf16)(a0.z * s); a[3]  = (__bf16)(a0.w * s);
    a[4]  = (__bf16)(a1.x * s); a[5]  = (__bf16)(a1.y * s);
    a[6]  = (__bf16)(a1.z * s); a[7]  = (__bf16)(a1.w * s);
    a[8]  = (__bf16)(a2.x * s); a[9]  = (__bf16)(a2.y * s);
    a[10] = (__bf16)(a2.z * s); a[11] = (__bf16)(a2.w * s);
    a[12] = (__bf16)(a3.x * s); a[13] = (__bf16)(a3.y * s);
    a[14] = (__bf16)(a3.z * s); a[15] = (__bf16)(a3.w * s);

    const v16bf* fb = fragv + (size_t)(kt * 8) * 32 + lane;
    v16bf b = fb[0];                   // double-buffered B fragments
    #pragma unroll
    for (int nt = 0; nt < 8; ++nt) {
      v16bf bn;
      if (nt < 7) bn = fb[(size_t)(nt + 1) * 32];
      acc[nt] = __builtin_amdgcn_wmma_f32_16x16x32_bf16(
          false, a, false, b, (short)0, acc[nt], false, false);
      b = bn;
    }
  }

  // Store: single wave-uniform bounds test; full strips take a branch-free path.
  if (rowbase + 16 <= N) {
    #pragma unroll
    for (int j = 0; j < 8; ++j) {
      float* crow = C + (size_t)(rowbase + h * 8 + j) * DH + rl;
      #pragma unroll
      for (int nt = 0; nt < 8; ++nt) {
        float v = acc[nt][j];
        if (EP == 1) v = tanhf(v);
        crow[nt * 16] = v;
      }
    }
  } else {
    #pragma unroll
    for (int j = 0; j < 8; ++j) {
      int r = rowbase + h * 8 + j;
      if (r < N) {
        float* crow = C + (size_t)r * DH + rl;
        #pragma unroll
        for (int nt = 0; nt < 8; ++nt) {
          float v = acc[nt][j];
          if (EP == 1) v = tanhf(v);
          crow[nt * 16] = v;
        }
      }
    }
  }
}

// y = [h0|h1|h2] @ concat_w.T  (K = 384, fragment-prepped transposed weight)
__global__ __launch_bounds__(256)
void gemm_concat_wmma(const float* __restrict__ A0, const float* __restrict__ A1,
                      const float* __restrict__ A2, const __bf16* __restrict__ frag,
                      float* __restrict__ C, int N) {
  const int wave = threadIdx.x >> 5, lane = threadIdx.x & 31;
  const int h = lane >> 4, rl = lane & 15;
  const int rowbase = (blockIdx.x * 8 + wave) * 16;
  int row = rowbase + rl;
  int rowc = row < N ? row : N - 1;
  const v16bf* fragv = (const v16bf*)frag;

  v8f acc[8] = {};
  for (int kt = 0; kt < 12; ++kt) {
    const float* ab = (kt < 4) ? A0 : ((kt < 8) ? A1 : A2);
    const float* arow = ab + (size_t)rowc * DH;
    int kb = ((kt & 3) << 5) + h * 8;
    float4 a0 = *(const float4*)(arow + kb);
    float4 a1 = *(const float4*)(arow + kb + 4);
    float4 a2 = *(const float4*)(arow + kb + 16);
    float4 a3 = *(const float4*)(arow + kb + 20);
    v16bf a;
    a[0]  = (__bf16)a0.x; a[1]  = (__bf16)a0.y; a[2]  = (__bf16)a0.z; a[3]  = (__bf16)a0.w;
    a[4]  = (__bf16)a1.x; a[5]  = (__bf16)a1.y; a[6]  = (__bf16)a1.z; a[7]  = (__bf16)a1.w;
    a[8]  = (__bf16)a2.x; a[9]  = (__bf16)a2.y; a[10] = (__bf16)a2.z; a[11] = (__bf16)a2.w;
    a[12] = (__bf16)a3.x; a[13] = (__bf16)a3.y; a[14] = (__bf16)a3.z; a[15] = (__bf16)a3.w;

    const v16bf* fb = fragv + (size_t)(kt * 8) * 32 + lane;
    v16bf b = fb[0];
    #pragma unroll
    for (int nt = 0; nt < 8; ++nt) {
      v16bf bn;
      if (nt < 7) bn = fb[(size_t)(nt + 1) * 32];
      acc[nt] = __builtin_amdgcn_wmma_f32_16x16x32_bf16(
          false, a, false, b, (short)0, acc[nt], false, false);
      b = bn;
    }
  }

  if (rowbase + 16 <= N) {
    #pragma unroll
    for (int j = 0; j < 8; ++j) {
      float* crow = C + (size_t)(rowbase + h * 8 + j) * DH + rl;
      #pragma unroll
      for (int nt = 0; nt < 8; ++nt) crow[nt * 16] = acc[nt][j];
    }
  } else {
    #pragma unroll
    for (int j = 0; j < 8; ++j) {
      int r = rowbase + h * 8 + j;
      if (r < N) {
        float* crow = C + (size_t)r * DH + rl;
        #pragma unroll
        for (int nt = 0; nt < 8; ++nt) crow[nt * 16] = acc[nt][j];
      }
    }
  }
}

// ---------------------------------------------------------------- scatter add
// agg[dst] += M[src]   (one wave per edge; 4 floats per lane; M/agg live in L2)
__global__ void scatter_add(const int* __restrict__ srcIdx, const int* __restrict__ dstIdx,
                            const float* __restrict__ M, float* __restrict__ agg, int E) {
  long long gid = (long long)blockIdx.x * blockDim.x + threadIdx.x;
  int e = (int)(gid >> 5);
  int lane = (int)(gid & 31);
  if (e >= E) return;
  int s = srcIdx[e], d = dstIdx[e];
  float4 v = *(const float4*)(M + (size_t)s * DH + lane * 4);
  float* base = agg + (size_t)d * DH + lane * 4;
  atomicAdd(base + 0, v.x);
  atomicAdd(base + 1, v.y);
  atomicAdd(base + 2, v.z);
  atomicAdd(base + 3, v.w);
}

// vals = vals * rsqrt(deg_in)[n] + b[d] + h[n,d]
__global__ void conv_epilogue(float* __restrict__ vals, const float* __restrict__ scaleIn,
                              const float* __restrict__ bias, const float* __restrict__ hres,
                              int N) {
  long long tid = (long long)blockIdx.x * blockDim.x + threadIdx.x;
  if (tid >= (long long)N * DH) return;
  int n = (int)(tid >> 7);
  int d = (int)(tid & 127);
  vals[tid] = vals[tid] * scaleIn[n] + bias[d] + hres[tid];
}

// ---------------------------------------------------------------- attention
// keys = tanh(vals @ attn_w); scores = keys @ attn_q; softmax over R; weighted sum.
__global__ __launch_bounds__(256)
void attn_agg(const float* __restrict__ v0, const float* __restrict__ v1,
              const float* __restrict__ v2, const float* __restrict__ aw,
              const float* __restrict__ aq, float* __restrict__ hout,
              float* __restrict__ attnOut, int N) {
  __shared__ float Wl[DH * 64];   // 32 KB
  __shared__ float Ql[64];
  for (int i = threadIdx.x; i < DH * 64; i += blockDim.x) Wl[i] = aw[i];
  if (threadIdx.x < 64) Ql[threadIdx.x] = aq[threadIdx.x];
  __syncthreads();

  int lane = threadIdx.x & 31;
  int wid  = blockIdx.x * (blockDim.x >> 5) + (threadIdx.x >> 5);
  int nw   = gridDim.x * (blockDim.x >> 5);
  const float* vs0 = v0; const float* vs1 = v1; const float* vs2 = v2;

  for (int n = wid; n < N; n += nw) {
    float4 vv[3];
    float sc[3];
    #pragma unroll
    for (int r = 0; r < 3; ++r) {
      const float* vr = (r == 0 ? vs0 : (r == 1 ? vs1 : vs2)) + (size_t)n * DH;
      float4 v = *(const float4*)(vr + lane * 4);
      vv[r] = v;
      float vj[4] = {v.x, v.y, v.z, v.w};
      float k0 = 0.0f, k1 = 0.0f;
      for (int sl = 0; sl < 32; ++sl) {
        #pragma unroll
        for (int j = 0; j < 4; ++j) {
          float vd = __shfl(vj[j], sl, 32);
          int d = sl * 4 + j;
          k0 = fmaf(vd, Wl[d * 64 + lane], k0);
          k1 = fmaf(vd, Wl[d * 64 + 32 + lane], k1);
        }
      }
      float t = tanhf(k0) * Ql[lane] + tanhf(k1) * Ql[32 + lane];
      #pragma unroll
      for (int off = 16; off > 0; off >>= 1) t += __shfl_xor(t, off, 32);
      sc[r] = t;
    }
    float m = fmaxf(sc[0], fmaxf(sc[1], sc[2]));
    float e0 = expf(sc[0] - m), e1 = expf(sc[1] - m), e2 = expf(sc[2] - m);
    float inv = 1.0f / (e0 + e1 + e2);
    float a0 = e0 * inv, a1 = e1 * inv, a2 = e2 * inv;
    float4 o;
    o.x = a0 * vv[0].x + a1 * vv[1].x + a2 * vv[2].x;
    o.y = a0 * vv[0].y + a1 * vv[1].y + a2 * vv[2].y;
    o.z = a0 * vv[0].z + a1 * vv[1].z + a2 * vv[2].z;
    o.w = a0 * vv[0].w + a1 * vv[1].w + a2 * vv[2].w;
    *(float4*)(hout + (size_t)n * DH + lane * 4) = o;
    if (attnOut && lane < 3)
      attnOut[(size_t)n * 3 + lane] = (lane == 0) ? a0 : ((lane == 1) ? a1 : a2);
  }
}

// ---------------------------------------------------------------- launch
extern "C" void kernel_launch(void* const* d_in, const int* in_sizes, int n_in,
                              void* d_out, int out_size, void* d_ws, size_t ws_size,
                              hipStream_t stream) {
  (void)in_sizes; (void)n_in; (void)out_size; (void)ws_size;
  const float* x        = (const float*)d_in[0];
  const int*   edges    = (const int*)  d_in[1];
  const float* feat_w   = (const float*)d_in[2];
  const float* conv_w   = (const float*)d_in[3];
  const float* conv_b   = (const float*)d_in[4];
  const float* attn_w   = (const float*)d_in[5];
  const float* attn_q   = (const float*)d_in[6];
  const float* concat_w = (const float*)d_in[7];
  float* out = (float*)d_out;

  const int N = NN, E = NE;
  const size_t NF = (size_t)N * DH;

  float* ws = (float*)d_ws;
  float* h0 = ws;
  float* h1 = ws + NF;
  float* hT = ws + 2 * NF;          // t, then reused as h2
  float* vals[3] = {ws + 3 * NF, ws + 4 * NF, ws + 5 * NF};
  float* Mb  = ws + 6 * NF;
  float* scO = ws + 7 * NF;                 // 3N rsqrt(deg_out) per relation
  float* scI = scO + (size_t)3 * N;         // 3N rsqrt(deg_in)
  __bf16* fragBase   = (__bf16*)(scI + (size_t)3 * N);
  __bf16* fragFeat   = fragBase;                               // 16384 bf16
  __bf16* fragConcat = fragBase + (size_t)16384 * 7;           // 49152 bf16

  // ---- weight fragment prep (bf16 WMMA B layout)
  prep_frag<<<64, 256, 0, stream>>>(feat_w, fragFeat, 128, 0);
  for (int l = 0; l < 2; ++l)
    for (int r = 0; r < 3; ++r)
      prep_frag<<<64, 256, 0, stream>>>(conv_w + (size_t)(l * 3 + r) * DH * DH,
                                        fragBase + (size_t)16384 * (1 + l * 3 + r), 128, 0);
  prep_frag<<<192, 256, 0, stream>>>(concat_w, fragConcat, 384, 1);

  // ---- degree scale factors (shared by all layers)
  zero_f32<<<(int)((6LL * N + 255) / 256), 256, 0, stream>>>(scO, 6LL * N);
  count_deg<<<(int)((3LL * E + 255) / 256), 256, 0, stream>>>(edges, scO, scI);
  finalize_deg<<<(int)((6LL * N + 255) / 256), 256, 0, stream>>>(scO, 6LL * N);

  const int gGemm = (N + 127) / 128;           // 391 blocks, 8 waves each
  const int gElem = (int)((NF + 255) / 256);
  const int gScat = (int)(((long long)E * 32 + 255) / 256);

  // h0 = tanh(x @ feat_w)
  gemm128_wmma<1><<<gGemm, 256, 0, stream>>>(x, nullptr, fragFeat, h0, N);

  auto hetLayer = [&](const float* hin, int l, float* hout, float* attnOut) {
    for (int r = 0; r < 3; ++r) {
      __bf16* fc = fragBase + (size_t)16384 * (1 + l * 3 + r);
      gemm128_wmma<0><<<gGemm, 256, 0, stream>>>(hin, scO + (size_t)r * N, fc, Mb, N);
      zero_f32<<<gElem, 256, 0, stream>>>(vals[r], (long long)NF);
      scatter_add<<<gScat, 256, 0, stream>>>(edges + (size_t)(r * 2 + 0) * E,
                                             edges + (size_t)(r * 2 + 1) * E,
                                             Mb, vals[r], E);
      conv_epilogue<<<gElem, 256, 0, stream>>>(vals[r], scI + (size_t)r * N,
                                               conv_b + (size_t)(l * 3 + r) * DH, hin, N);
    }
    attn_agg<<<512, 256, 0, stream>>>(vals[0], vals[1], vals[2],
                                      attn_w + (size_t)l * DH * 64,
                                      attn_q + (size_t)l * 64, hout, attnOut, N);
  };

  hetLayer(h0, 1, h1, nullptr);        // h1
  hetLayer(h0, 0, hT, nullptr);        // t
  hetLayer(hT, 1, hT, out + NF);       // h2 (reuses t buffer) + attn.T -> d_out tail
  gemm_concat_wmma<<<gGemm, 256, 0, stream>>>(h0, h1, hT, fragConcat, out, N);
}
